// GCN_27659589386355
// MI455X (gfx1250) — matrix operand order repaired and checked by reading, not compiled, not run on hardware
//
#include <hip/hip_runtime.h>
#include <hip/hip_bf16.h>

typedef __attribute__((ext_vector_type(16))) _Float16 v16h;
typedef __attribute__((ext_vector_type(8)))  _Float16 v8h;
typedef __attribute__((ext_vector_type(8)))  float    v8f;
typedef __attribute__((ext_vector_type(4)))  unsigned u32x4;
typedef __attribute__((ext_vector_type(8)))  int      i32x8;
typedef __attribute__((ext_vector_type(4)))  int      i32x4;

#define TB 256
#define WPB 8   // waves per block (wave32)

// ---------------- degree / normalization ----------------
__global__ void k_init_deg(float* deg, int n) {
  int i = blockIdx.x * blockDim.x + threadIdx.x;
  if (i < n) deg[i] = 1.0f;  // self loop contributes 1
}

__global__ void k_count_deg(float* deg, const int* __restrict__ dst, int E) {
  int e = blockIdx.x * blockDim.x + threadIdx.x;
  if (e < E) unsafeAtomicAdd(&deg[dst[e]], 1.0f);
}

__global__ void k_rsqrt_inplace(float* d, int n) {
  int i = blockIdx.x * blockDim.x + threadIdx.x;
  if (i < n) d[i] = rsqrtf(d[i]);
}

// ---------------- dtype conversion ----------------
__global__ void k_f32_to_f16(const float* __restrict__ in, _Float16* __restrict__ out, size_t n) {
  size_t i = (size_t)blockIdx.x * blockDim.x + threadIdx.x;
  if (i < n) out[i] = (_Float16)in[i];
}

// W [K,C] row-major f32 -> Wt [C,K] row-major f16 (i.e., column-major of W)
__global__ void k_transpose_to_f16(const float* __restrict__ W, _Float16* __restrict__ Wt,
                                   int K, int C) {
  int i = blockIdx.x * blockDim.x + threadIdx.x;
  if (i < K * C) {
    int k = i / C, c = i % C;
    Wt[(size_t)c * K + k] = (_Float16)W[i];
  }
}

// ---------------- WMMA GEMM: D[M,C] = A[M,K](f16) * B[K,C] (Bt is [C,K] f16) ----------------
// A-slab (16 rows x K f16) staged into LDS by the Tensor Data Mover (one
// tensor_load_to_lds per block, tracked with TENSORcnt), then 8 waves run the
// v_wmma_f32_16x16x32_f16 K-loop with B fragments streamed from L2.
__global__ __launch_bounds__(TB) void k_gemm_wmma(
    const _Float16* __restrict__ A, const _Float16* __restrict__ Bt,
    float* __restrict__ D, int M, int K, int C) {
  __shared__ _Float16 As[16 * 512];  // 16 rows x K (K<=512) = 16KB max; LDS offset 0

  const int tid   = threadIdx.x;
  const int mbase = blockIdx.x * 16;

  // ---- TDM: DMA the 16xK f16 tile (row stride K) into LDS, zero-fill OOB rows
  if (tid < 32) {
    unsigned long long gaddr = (unsigned long long)(A + (size_t)mbase * K);
    int rows = M - mbase; if (rows > 16) rows = 16;   // tensor rows left (OOB -> 0)

    u32x4 g0;
    g0[0] = 1u;                                        // count=1, is_restore=0, no gather
    g0[1] = 0u;                                        // lds_addr = 0 (As at LDS base)
    g0[2] = (unsigned)(gaddr & 0xffffffffu);           // global_addr[31:0]
    g0[3] = (unsigned)((gaddr >> 32) & 0x1ffffffu) | (2u << 30);  // addr[56:32], type=2

    i32x8 g1;
    g1[0] = 1 << 16;                                   // wg_mask=0, data_size=1 (2 bytes)
    g1[1] = (K & 0xffff) << 16;                        // tensor_dim0[15:0]
    g1[2] = ((K >> 16) & 0xffff) | ((rows & 0xffff) << 16);  // td0[31:16] | td1[15:0]
    g1[3] = ((K & 0xffff) << 16);                      // td1[31:16]=0 | tile_dim0=K
    g1[4] = 16;                                        // tile_dim1=16, tile_dim2=0
    g1[5] = K;                                         // tensor_dim0_stride[31:0]
    g1[6] = 0;                                         // stride hi bits / dim1_stride
    g1[7] = 0;

    i32x4 gz4 = {0, 0, 0, 0};                          // 2-D tensor: groups 2/3 unused
    i32x8 gz8 = {0, 0, 0, 0, 0, 0, 0, 0};
    __builtin_amdgcn_tensor_load_to_lds(g0, g1, gz4, gz4, gz8, 0);
    __builtin_amdgcn_s_wait_tensorcnt(0);
  }
  // The TDM engine wrote As behind the compiler's back (the builtin carries the
  // LDS address only as integer descriptor data). Escape As and clobber memory
  // so the ds_load fragment reads below are not folded away as undef.
  {
    _Float16* asp = &As[0];
    asm volatile("" : "+v"(asp) :: "memory");
  }
  __syncthreads();

  const int wave = tid >> 5;
  const int lane = tid & 31;
  const int hi   = lane >> 4;   // half-wave select
  const int lo   = lane & 15;   // M (for A) / N (for B)

  for (int nt = wave; nt * 16 < C; nt += WPB) {
    v8f acc = {};
    const _Float16* bcol = Bt + (size_t)(nt * 16 + lo) * K;

    for (int k0 = 0; k0 < K; k0 += 32) {
      // A fragment (16x32 f16): lane lo = row M; elems 0..7 -> K=k0+8*hi..+7,
      // elems 8..15 -> K=k0+16+8*hi..+7  (ISA 16-bit A layout)
      v8h alo = *(const v8h*)&As[lo * K + k0 + 8 * hi];
      v8h ahi = *(const v8h*)&As[lo * K + k0 + 16 + 8 * hi];
      v16h a;
#pragma unroll
      for (int i = 0; i < 8; ++i) { a[i] = alo[i]; a[8 + i] = ahi[i]; }

      // B fragment (32x16 f16): lane lo = col N; elems 0..7 -> K=k0+8*hi..+7,
      // elems 8..15 -> K=k0+16+8*hi..+7 (matches A)
      v8h blo = *(const v8h*)&bcol[k0 + 8 * hi];
      v8h bhi = *(const v8h*)&bcol[k0 + 16 + 8 * hi];
      v16h b;
#pragma unroll
      for (int i = 0; i < 8; ++i) { b[i] = blo[i]; b[8 + i] = bhi[i]; }

      acc = __builtin_amdgcn_wmma_f32_16x16x32_f16(
          /*neg_a=*/false, a, /*neg_b=*/false, b,
          /*c_mod=*/(short)0, acc, /*reuse_a=*/false, /*reuse_b=*/false);
    }

    // C/D layout: VGPR v, lane -> row = v + 8*hi, col = lo
#pragma unroll
    for (int v = 0; v < 8; ++v) {
      int row = mbase + v + 8 * hi;
      if (row < M) D[(size_t)row * C + nt * 16 + lo] = acc[v];
    }
  }
}

// ---------------- aggregation ----------------
// agg[i,c] = t[i,c] * dinv[i]^2 (+ bias[c]) : self-loop term + optional fused bias
__global__ void k_selfloop_init(const float* __restrict__ t, const float* __restrict__ dinv,
                                const float* __restrict__ bias, float* __restrict__ agg,
                                int Nrows, int C) {
  size_t i = (size_t)blockIdx.x * blockDim.x + threadIdx.x;
  size_t total = (size_t)Nrows * C;
  if (i < total) {
    int r = (int)(i / C), c = (int)(i % C);
    float dv = dinv[r];
    float b = bias ? bias[c] : 0.0f;
    agg[i] = t[i] * dv * dv + b;
  }
}

// one wave per edge: gather row t[src], scale by dinv[src]*dinv[dst], atomic scatter to agg[dst]
__global__ __launch_bounds__(TB) void k_edge_agg(
    const float* __restrict__ t, const int* __restrict__ src, const int* __restrict__ dst,
    const float* __restrict__ dinv, float* agg, int E, int C) {
  int e = blockIdx.x * WPB + (threadIdx.x >> 5);
  if (e >= E) return;
  int s = src[e], d = dst[e];
  float nrm = dinv[s] * dinv[d];
  int lane = threadIdx.x & 31;
  const float4* ts = (const float4*)(t + (size_t)s * C);
  float* ad = agg + (size_t)d * C;
  for (int c4 = lane; c4 * 4 < C; c4 += 32) {
    float4 v = ts[c4];
    unsafeAtomicAdd(&ad[c4 * 4 + 0], v.x * nrm);
    unsafeAtomicAdd(&ad[c4 * 4 + 1], v.y * nrm);
    unsafeAtomicAdd(&ad[c4 * 4 + 2], v.z * nrm);
    unsafeAtomicAdd(&ad[c4 * 4 + 3], v.w * nrm);
  }
}

__global__ void k_bias_relu_f16(const float* __restrict__ agg, const float* __restrict__ b,
                                _Float16* __restrict__ h16, int Nrows, int C) {
  size_t i = (size_t)blockIdx.x * blockDim.x + threadIdx.x;
  size_t total = (size_t)Nrows * C;
  if (i < total) {
    int c = (int)(i % C);
    float v = agg[i] + b[c];
    h16[i] = (_Float16)fmaxf(v, 0.0f);
  }
}

// ---------------- log-softmax over 64 classes, one wave per row ----------------
__global__ __launch_bounds__(TB) void k_log_softmax64(float* out, int Nrows) {
  int row = blockIdx.x * WPB + (threadIdx.x >> 5);
  if (row >= Nrows) return;
  int lane = threadIdx.x & 31;
  float* p = out + (size_t)row * 64;
  float v0 = p[lane], v1 = p[lane + 32];
  float mx = fmaxf(v0, v1);
#pragma unroll
  for (int off = 16; off > 0; off >>= 1) mx = fmaxf(mx, __shfl_xor(mx, off));
  float se = expf(v0 - mx) + expf(v1 - mx);
#pragma unroll
  for (int off = 16; off > 0; off >>= 1) se += __shfl_xor(se, off);
  float lse = mx + logf(se);
  p[lane]      = v0 - lse;
  p[lane + 32] = v1 - lse;
}

// ---------------- host orchestration ----------------
extern "C" void kernel_launch(void* const* d_in, const int* in_sizes, int n_in,
                              void* d_out, int out_size, void* d_ws, size_t ws_size,
                              hipStream_t stream) {
  const float* x  = (const float*)d_in[0];
  const int*   ei = (const int*)d_in[1];
  const float* W1 = (const float*)d_in[2];
  const float* b1 = (const float*)d_in[3];
  const float* W2 = (const float*)d_in[4];
  const float* b2 = (const float*)d_in[5];
  float* out = (float*)d_out;

  const int FIN = 512, FH = 256, FO = 64;
  const int N = in_sizes[0] / FIN;
  const int E = in_sizes[1] / 2;
  const int* src = ei;
  const int* dst = ei + E;

  // workspace carve-out (256B aligned)
  char* ws = (char*)d_ws;
  size_t off = 0;
  auto carve = [&](size_t bytes) -> char* {
    char* p = ws + off;
    off = (off + bytes + 255) & ~(size_t)255;
    return p;
  };
  float*    dinv = (float*)carve((size_t)N * 4);            // deg then dinv in place
  _Float16* W1t  = (_Float16*)carve((size_t)FIN * FH * 2);  // [FH,FIN] col-major of W1
  _Float16* W2t  = (_Float16*)carve((size_t)FH * FO * 2);   // [FO,FH]
  _Float16* x16  = (_Float16*)carve((size_t)N * FIN * 2);   // f16 features
  float*    t1   = (float*)carve((size_t)N * FH * 4);       // x @ W1
  float*    agg1 = (float*)carve((size_t)N * FH * 4);       // aggregated layer-1
  // aliases into the (dead after GEMM1) x16 region:
  _Float16* h16 = x16;                                      // relu(agg1+b1) in f16: N*FH*2
  float*    t2  = (float*)((char*)x16 + (size_t)N * FH * 2);// h @ W2 : N*FO*4

  // 1) degree + dinv
  k_init_deg<<<(N + TB - 1) / TB, TB, 0, stream>>>(dinv, N);
  k_count_deg<<<(E + TB - 1) / TB, TB, 0, stream>>>(dinv, dst, E);
  k_rsqrt_inplace<<<(N + TB - 1) / TB, TB, 0, stream>>>(dinv, N);

  // 2) precision conversion / weight transpose
  {
    size_t n = (size_t)N * FIN;
    k_f32_to_f16<<<(unsigned)((n + TB - 1) / TB), TB, 0, stream>>>(x, x16, n);
  }
  k_transpose_to_f16<<<(FIN * FH + TB - 1) / TB, TB, 0, stream>>>(W1, W1t, FIN, FH);
  k_transpose_to_f16<<<(FH * FO + TB - 1) / TB, TB, 0, stream>>>(W2, W2t, FH, FO);

  // 3) layer 1: GEMM + aggregate + bias/ReLU (to f16)
  k_gemm_wmma<<<(N + 15) / 16, TB, 0, stream>>>(x16, W1t, t1, N, FIN, FH);
  {
    size_t n = (size_t)N * FH;
    k_selfloop_init<<<(unsigned)((n + TB - 1) / TB), TB, 0, stream>>>(t1, dinv, nullptr, agg1, N, FH);
  }
  k_edge_agg<<<(E + WPB - 1) / WPB, TB, 0, stream>>>(t1, src, dst, dinv, agg1, E, FH);
  {
    size_t n = (size_t)N * FH;
    k_bias_relu_f16<<<(unsigned)((n + TB - 1) / TB), TB, 0, stream>>>(agg1, b1, h16, N, FH);
  }

  // 4) layer 2: GEMM + aggregate (+bias fused into init) -> d_out
  k_gemm_wmma<<<(N + 15) / 16, TB, 0, stream>>>(h16, W2t, t2, N, FH, FO);
  {
    size_t n = (size_t)N * FO;
    k_selfloop_init<<<(unsigned)((n + TB - 1) / TB), TB, 0, stream>>>(t2, dinv, b2, out, N, FO);
  }
  k_edge_agg<<<(E + WPB - 1) / WPB, TB, 0, stream>>>(t2, src, dst, dinv, out, E, FO);

  // 5) log-softmax over 64 classes
  k_log_softmax64<<<(N + WPB - 1) / WPB, TB, 0, stream>>>(out, N);
}
